// ActorCriticMFOSRNN_62843961475533
// MI455X (gfx1250) — matrix operand order repaired and checked by reading, not compiled, not run on hardware
//
#include <hip/hip_runtime.h>

// ---------------------------------------------------------------------------
// ActorCritic MFOS-RNN for MI455X (gfx1250, wave32, WMMA bf16 16x16x32).
//
// Pipeline (all on `stream`):
//   k_wc   : Wc = W_emb @ Wi  (f32 dot, per head) -> bf16, B-fragment swizzled
//   k_bc   : bc = b_emb @ Wi + bi (f32)
//   k_swz  : Whrz/Whn/W_th f32 -> bf16 B-fragment swizzled
//   k_obs  : obs f32 -> bf16 A-fragment swizzled
//   k_scan : 96 wgs x 128 thr; per wg: one head + 16 batch rows, 256 steps.
//            All 36 weight B-fragments (288 VGPRs) held in registers across
//            the whole scan (gfx1250 1024-VGPR file). obs fragments + done
//            bytes are software-pipelined one step ahead so the sequential
//            loop never stalls on global latency. Per step/wave:
//            36 v_wmma_f32_16x16x32_bf16 + 8 ds_load_b128 + gates.
//   k_th   : th = sigmoid(y_t @ W_th + b_th); W_th fragments persistent in
//            registers, 8 row-tiles per wave.
//   k_ac   : actor logits (N=6) + critic (N=1) VALU post-pass
//
// Workspace layout (bytes), total ~118.5 MB:
//   [0)        Wc swz      3*49152
//   [147456)   Whrz swz    3*65536
//   [344064)   Whn swz     3*32768
//   [442368)   Wth swz     32768
//   [475136)   bc          3*384*4
//   [524288)   obs swz     16777216
//   [17825792) y swz       3*33554432
// ---------------------------------------------------------------------------

#define T_ 256
#define B_ 512
#define OBS_ 64
#define H_ 128
#define A_ 6

typedef __bf16 v16bf __attribute__((ext_vector_type(16)));
typedef float  v8f   __attribute__((ext_vector_type(8)));
typedef unsigned int v4u __attribute__((ext_vector_type(4)));
typedef unsigned int v2u __attribute__((ext_vector_type(2)));

union WFrag {
    v16bf v;
    v4u   q[2];
    unsigned short h[16];
};

__device__ __forceinline__ unsigned short f2bf(float f) {
    unsigned u = __float_as_uint(f);
    unsigned r = u + 0x7fffu + ((u >> 16) & 1u);   // round-to-nearest-even
    return (unsigned short)(r >> 16);
}
__device__ __forceinline__ float bf2f(unsigned short h) {
    return __uint_as_float(((unsigned)h) << 16);
}
__device__ __forceinline__ float sigm(float x) {
    return __builtin_amdgcn_rcpf(1.0f + __expf(-x));
}
__device__ __forceinline__ float tanh_f(float x) {
    float xx = fminf(fmaxf(x, -15.0f), 15.0f);
    float e = __expf(2.0f * xx);
    return (e - 1.0f) * __builtin_amdgcn_rcpf(e + 1.0f);
}
__device__ __forceinline__ v8f wmma_bf16(v16bf a, v16bf b, v8f c) {
    return __builtin_amdgcn_wmma_f32_16x16x32_bf16(false, a, false, b,
                                                   (short)0, c, false, false);
}
__device__ __forceinline__ v16bf ld_frag(const unsigned short* p) {
    WFrag f;
    f.q[0] = *(const v4u*)p;
    f.q[1] = *(const v4u*)(p + 8);
    return f.v;
}

// ---------------- prep: Wc = W_emb @ Wi, swizzled bf16 (KB=2, NB=24) --------
__global__ void k_wc(const float* __restrict__ We_t, const float* __restrict__ We_a,
                     const float* __restrict__ We_c,
                     const float* __restrict__ Wi_t, const float* __restrict__ Wi_a,
                     const float* __restrict__ Wi_c,
                     unsigned short* __restrict__ dst) {
    int idx = blockIdx.x * 256 + threadIdx.x;
    if (idx >= 3 * 64 * 384) return;
    int hd = idx / (64 * 384), rem = idx % (64 * 384);
    int k = rem / 384, n = rem % 384;
    const float* We = hd == 0 ? We_t : (hd == 1 ? We_a : We_c);
    const float* Wi = hd == 0 ? Wi_t : (hd == 1 ? Wi_a : Wi_c);
    float s = 0.f;
    for (int m = 0; m < 128; m++) s += We[k * 128 + m] * Wi[m * 384 + n];
    int kb = k >> 5, kr = k & 31, nb = n >> 4, nl = n & 15;
    int lane = nl + (((kr >> 3) & 1) << 4);
    dst[hd * 24576 + ((nb * 2 + kb) * 32 + lane) * 16 + ((kr & 7) + ((kr >> 4) << 3))] = f2bf(s);
}

// ---------------- prep: bc = b_emb @ Wi + bi --------------------------------
__global__ void k_bc(const float* __restrict__ be_t, const float* __restrict__ be_a,
                     const float* __restrict__ be_c,
                     const float* __restrict__ Wi_t, const float* __restrict__ Wi_a,
                     const float* __restrict__ Wi_c,
                     const float* __restrict__ bi_t, const float* __restrict__ bi_a,
                     const float* __restrict__ bi_c,
                     float* __restrict__ bc) {
    int e = blockIdx.x * 256 + threadIdx.x;
    if (e >= 3 * 384) return;
    int hd = e / 384, n = e % 384;
    const float* be = hd == 0 ? be_t : (hd == 1 ? be_a : be_c);
    const float* Wi = hd == 0 ? Wi_t : (hd == 1 ? Wi_a : Wi_c);
    const float* bi = hd == 0 ? bi_t : (hd == 1 ? bi_a : bi_c);
    float s = bi[n];
    for (int m = 0; m < 128; m++) s += be[m] * Wi[m * 384 + n];
    bc[e] = s;
}

// ---------------- prep: generic f32[K][N] -> bf16 B-fragment swizzle --------
__global__ void k_swz(const float* __restrict__ src, unsigned short* __restrict__ dst,
                      int K, int N) {
    int e = blockIdx.x * 256 + threadIdx.x;
    if (e >= K * N) return;
    int k = e / N, n = e % N;
    int kb = k >> 5, kr = k & 31, nb = n >> 4, nl = n & 15;
    int lane = nl + (((kr >> 3) & 1) << 4);
    int KB = K >> 5;
    dst[((nb * KB + kb) * 32 + lane) * 16 + ((kr & 7) + ((kr >> 4) << 3))] = f2bf(src[(size_t)k * N + n]);
}

// ---------------- prep: obs f32 -> bf16 A-fragment swizzle ------------------
__global__ void k_obs(const float* __restrict__ src, unsigned short* __restrict__ dst) {
    int idx = blockIdx.x * 256 + threadIdx.x;            // 524288 chunks
    int lane = idx & 31, kb = (idx >> 5) & 1, tile = (idx >> 6) & 31, t = idx >> 11;
    int m = lane & 15, hi = lane >> 4;
    int kbase = kb * 32 + hi * 8;
    const float* s = src + ((size_t)(t * B_ + tile * 16 + m) * OBS_ + kbase);
    WFrag f;
#pragma unroll
    for (int i = 0; i < 8; i++) { f.h[i] = f2bf(s[i]); f.h[8 + i] = f2bf(s[16 + i]); }
    unsigned short* d = dst + ((size_t)(t * 32 + tile) * 2 + kb) * 512 + lane * 16;
    *(v4u*)d = f.q[0];
    *(v4u*)(d + 8) = f.q[1];
}

// ---------------- the GRU scan ----------------------------------------------
__global__ __launch_bounds__(128, 1) void k_scan(
    const float* __restrict__ hidden, const unsigned char* __restrict__ dones,
    const float* __restrict__ bhn_t, const float* __restrict__ bhn_a,
    const float* __restrict__ bhn_c,
    const unsigned short* __restrict__ wWc, const unsigned short* __restrict__ wWhrz,
    const unsigned short* __restrict__ wWhn, const float* __restrict__ wBc,
    const unsigned short* __restrict__ wObs, unsigned short* __restrict__ wY,
    float* __restrict__ outHid) {
    __shared__ __attribute__((aligned(16))) unsigned short hb[2][16 * 136];

    const int tid = threadIdx.x;
    const int w = tid >> 5, lane = tid & 31, m = lane & 15, hi = lane >> 4;
    const int head = blockIdx.x >> 5;          // 0..2
    const int tile = blockIdx.x & 31;          // 0..31
    const int tb = tile * 16;

    const unsigned short* Wc   = wWc   + head * 24576;
    const unsigned short* Whrz = wWhrz + head * 32768;
    const unsigned short* Whn  = wWhn  + head * 16384;
    const float* bhn = (head == 0) ? bhn_t : ((head == 1) ? bhn_a : bhn_c);

    float bc_[3][2], bh_[2];
#pragma unroll
    for (int s = 0; s < 3; s++)
#pragma unroll
        for (int c = 0; c < 2; c++)
            bc_[s][c] = wBc[head * 384 + s * 128 + w * 32 + c * 16 + m];
#pragma unroll
    for (int c = 0; c < 2; c++) bh_[c] = bhn[w * 32 + c * 16 + m];

    // ---- hoist ALL weight B-fragments for this wave's N-slice into VGPRs ---
    // 36 fragments x 8 VGPRs = 288 VGPRs, persistent across all 256 steps.
    v16bf bWc[2][6];     // [kb][s*2+c]
    v16bf bRZ[4][4];     // [kb][s*2+c]
    v16bf bNN[4][2];     // [kb][c]
#pragma unroll
    for (int kb = 0; kb < 2; kb++)
#pragma unroll
        for (int s = 0; s < 3; s++)
#pragma unroll
            for (int c = 0; c < 2; c++) {
                const int nb = s * 8 + w * 2 + c;
                bWc[kb][s * 2 + c] = ld_frag(Wc + (nb * 2 + kb) * 512 + lane * 16);
            }
#pragma unroll
    for (int kb = 0; kb < 4; kb++) {
#pragma unroll
        for (int s = 0; s < 2; s++)
#pragma unroll
            for (int c = 0; c < 2; c++) {
                const int nb = s * 8 + w * 2 + c;
                bRZ[kb][s * 2 + c] = ld_frag(Whrz + (nb * 4 + kb) * 512 + lane * 16);
            }
#pragma unroll
        for (int c = 0; c < 2; c++) {
            const int nb = w * 2 + c;
            bNN[kb][c] = ld_frag(Whn + (nb * 4 + kb) * 512 + lane * 16);
        }
    }

    // h0 from `hidden` (split: head slice of 128 cols)
    float ho[2][8];
#pragma unroll
    for (int c = 0; c < 2; c++)
#pragma unroll
        for (int r = 0; r < 8; r++) {
            ho[c][r] = hidden[(size_t)(tb + r + 8 * hi) * 384 + head * 128 +
                              w * 32 + c * 16 + m];
            hb[0][(r + 8 * hi) * 136 + w * 32 + c * 16 + m] = f2bf(ho[c][r]);
        }
    __syncthreads();

    const v8f vz = {0.f, 0.f, 0.f, 0.f, 0.f, 0.f, 0.f, 0.f};
    const v4u qz = {0u, 0u, 0u, 0u};

    // ---- software-pipelined obs fragments + done bytes (one step ahead) ----
    WFrag pA0, pA1;
    v2u pdb;
    unsigned int pdown;
    {
        const unsigned short* op = wObs + ((size_t)tile * 2) * 512 + lane * 16;
        pA0.q[0] = *(const v4u*)op;        pA0.q[1] = *(const v4u*)(op + 8);
        pA1.q[0] = *(const v4u*)(op + 512); pA1.q[1] = *(const v4u*)(op + 520);
        const unsigned char* dn = dones + tb;
        pdown = dn[m];
        pdb = *(const v2u*)(dn + 8 * hi);
    }

    for (int t = 0; t < T_; t++) {
        const int rb = t & 1, wb = rb ^ 1;

        // consume prefetched values
        const WFrag a0 = pA0, a1 = pA1;
        const v2u db = pdb;
        const bool down = pdown != 0;

        // done[t] reset (applied to carry before use)
#pragma unroll
        for (int r = 0; r < 4; r++) {
            if ((db.x >> (8 * r)) & 0xffu) { ho[0][r] = 0.f; ho[1][r] = 0.f; }
            if ((db.y >> (8 * r)) & 0xffu) { ho[0][r + 4] = 0.f; ho[1][r + 4] = 0.f; }
        }

        v8f xp[3][2], rz[2][2], nng[2];
#pragma unroll
        for (int s = 0; s < 3; s++)
#pragma unroll
            for (int c = 0; c < 2; c++) xp[s][c] = vz;
#pragma unroll
        for (int s = 0; s < 2; s++)
#pragma unroll
            for (int c = 0; c < 2; c++) rz[s][c] = vz;
        nng[0] = vz; nng[1] = vz;

        // xp = obs @ Wc   (KB=2), A fragments already in registers
#pragma unroll
        for (int s = 0; s < 3; s++)
#pragma unroll
            for (int c = 0; c < 2; c++) {
                xp[s][c] = wmma_bf16(a0.v, bWc[0][s * 2 + c], xp[s][c]);
                xp[s][c] = wmma_bf16(a1.v, bWc[1][s * 2 + c], xp[s][c]);
            }

        // prefetch obs + done for t+1 (wraps harmlessly at t=255)
        {
            const int tn = (t + 1) & (T_ - 1);
            const unsigned short* op =
                wObs + ((size_t)(tn * 32 + tile) * 2) * 512 + lane * 16;
            pA0.q[0] = *(const v4u*)op;         pA0.q[1] = *(const v4u*)(op + 8);
            pA1.q[0] = *(const v4u*)(op + 512); pA1.q[1] = *(const v4u*)(op + 520);
            const unsigned char* dn = dones + tn * B_ + tb;
            pdown = dn[m];
            pdb = *(const v2u*)(dn + 8 * hi);
        }

        // hrz = h @ Whrz,  hn = h @ Whn   (KB=4)
#pragma unroll
        for (int kb = 0; kb < 4; kb++) {
            WFrag a;
            const int kbase = kb * 32 + hi * 8;
            a.q[0] = *(const v4u*)&hb[rb][m * 136 + kbase];
            a.q[1] = *(const v4u*)&hb[rb][m * 136 + kbase + 16];
            if (down) { a.q[0] = qz; a.q[1] = qz; }
#pragma unroll
            for (int s = 0; s < 2; s++)
#pragma unroll
                for (int c = 0; c < 2; c++)
                    rz[s][c] = wmma_bf16(a.v, bRZ[kb][s * 2 + c], rz[s][c]);
#pragma unroll
            for (int c = 0; c < 2; c++)
                nng[c] = wmma_bf16(a.v, bNN[kb][c], nng[c]);
        }

        // gates + state update, write h_new (bf16) into the other buffer
#pragma unroll
        for (int c = 0; c < 2; c++) {
            const int col = w * 32 + c * 16 + m;
#pragma unroll
            for (int r = 0; r < 8; r++) {
                const float rr = sigm(xp[0][c][r] + bc_[0][c] + rz[0][c][r]);
                const float zq = sigm(xp[1][c][r] + bc_[1][c] + rz[1][c][r]);
                const float nn = tanh_f(xp[2][c][r] + bc_[2][c] +
                                        rr * (nng[c][r] + bh_[c]));
                const float h2 = (1.0f - zq) * nn + zq * ho[c][r];
                ho[c][r] = h2;
                hb[wb][(r + 8 * hi) * 136 + col] = f2bf(h2);
            }
        }
        __syncthreads();

        // dump y (= h_new) A-fragment-swizzled to ws (wave w handles kb = w)
        {
            const int kbase = w * 32 + hi * 8;
            const v4u q0 = *(const v4u*)&hb[wb][m * 136 + kbase];
            const v4u q1 = *(const v4u*)&hb[wb][m * 136 + kbase + 16];
            unsigned short* yp =
                wY + ((size_t)((head * T_ + t) * 32 + tile) * 4 + w) * 512 + lane * 16;
            *(v4u*)yp = q0;
            *(v4u*)(yp + 8) = q1;
        }
    }

    // final hidden_out = concat(h_t, h_a, h_c)
#pragma unroll
    for (int c = 0; c < 2; c++)
#pragma unroll
        for (int r = 0; r < 8; r++)
            outHid[(size_t)(tb + r + 8 * hi) * 384 + head * 128 + w * 32 + c * 16 + m] =
                ho[c][r];
}

// ---------------- post: th = sigmoid(y_t @ W_th + b_th) ---------------------
// W_th B-fragments persistent in 256 VGPRs; each wave does 8 row-tiles.
__global__ __launch_bounds__(128, 1) void k_th(const unsigned short* __restrict__ yT,
                                               const unsigned short* __restrict__ Wth,
                                               const float* __restrict__ b_th,
                                               float* __restrict__ outTh) {
    const int w = threadIdx.x >> 5, lane = threadIdx.x & 31, m = lane & 15,
              hi = lane >> 4;
    const int tile0 = (blockIdx.x * 4 + w) * 8;   // 8 tiles per wave

    v16bf bW[4][8];   // [kb][nb]
#pragma unroll
    for (int kb = 0; kb < 4; kb++)
#pragma unroll
        for (int nb = 0; nb < 8; nb++)
            bW[kb][nb] = ld_frag(Wth + (nb * 4 + kb) * 512 + lane * 16);

    float bt[8];
#pragma unroll
    for (int nb = 0; nb < 8; nb++) bt[nb] = b_th[nb * 16 + m];

    const v8f vz = {0.f, 0.f, 0.f, 0.f, 0.f, 0.f, 0.f, 0.f};
    for (int i = 0; i < 8; i++) {
        const int rowTile = tile0 + i;
        v8f acc[8];
#pragma unroll
        for (int nb = 0; nb < 8; nb++) acc[nb] = vz;
#pragma unroll
        for (int kb = 0; kb < 4; kb++) {
            WFrag a;
            const unsigned short* ap =
                yT + ((size_t)rowTile * 4 + kb) * 512 + lane * 16;
            a.q[0] = *(const v4u*)ap;
            a.q[1] = *(const v4u*)(ap + 8);
#pragma unroll
            for (int nb = 0; nb < 8; nb++)
                acc[nb] = wmma_bf16(a.v, bW[kb][nb], acc[nb]);
        }
#pragma unroll
        for (int nb = 0; nb < 8; nb++) {
#pragma unroll
            for (int r = 0; r < 8; r++)
                outTh[(size_t)(rowTile * 16 + r + 8 * hi) * 128 + nb * 16 + m] =
                    sigm(acc[nb][r] + bt[nb]);
        }
    }
}

// ---------------- post: actor logits + critic (VALU, tiny N) ----------------
__global__ __launch_bounds__(256, 1) void k_ac(
    const unsigned short* __restrict__ yA, const unsigned short* __restrict__ yC,
    const float* __restrict__ W_act, const float* __restrict__ b_act,
    const float* __restrict__ W_crit, const float* __restrict__ b_crit,
    float* __restrict__ outAct, float* __restrict__ outCrit) {
    __shared__ float sWa[128 * 6];
    __shared__ float sWc[128];
    const int tid = threadIdx.x;
    for (int i = tid; i < 768; i += 256) sWa[i] = W_act[i];
    if (tid < 128) sWc[tid] = W_crit[tid];
    __syncthreads();

    const int row = blockIdx.x * 256 + tid;      // 0..131071
    const int tile = row >> 4, m = row & 15;
    float aa[6] = {0.f, 0.f, 0.f, 0.f, 0.f, 0.f};
    float cc = 0.f;
#pragma unroll
    for (int kb = 0; kb < 4; kb++) {
        const size_t base = ((size_t)tile * 4 + kb) * 512 + m * 16;
        WFrag a0, a1, c0, c1;
        a0.q[0] = *(const v4u*)&yA[base];       a0.q[1] = *(const v4u*)&yA[base + 8];
        a1.q[0] = *(const v4u*)&yA[base + 256]; a1.q[1] = *(const v4u*)&yA[base + 264];
        c0.q[0] = *(const v4u*)&yC[base];       c0.q[1] = *(const v4u*)&yC[base + 8];
        c1.q[0] = *(const v4u*)&yC[base + 256]; c1.q[1] = *(const v4u*)&yC[base + 264];
#pragma unroll
        for (int i = 0; i < 8; i++) {
            const float ya[4] = {bf2f(a0.h[i]), bf2f(a1.h[i]), bf2f(a0.h[8 + i]),
                                 bf2f(a1.h[8 + i])};
            const float yc[4] = {bf2f(c0.h[i]), bf2f(c1.h[i]), bf2f(c0.h[8 + i]),
                                 bf2f(c1.h[8 + i])};
            const int kk[4] = {kb * 32 + i, kb * 32 + 8 + i, kb * 32 + 16 + i,
                               kb * 32 + 24 + i};
#pragma unroll
            for (int j = 0; j < 4; j++) {
#pragma unroll
                for (int o = 0; o < 6; o++) aa[o] += ya[j] * sWa[kk[j] * 6 + o];
                cc += yc[j] * sWc[kk[j]];
            }
        }
    }
#pragma unroll
    for (int o = 0; o < 6; o++) outAct[(size_t)row * 6 + o] = aa[o] + b_act[o];
    outCrit[row] = cc + b_crit[0];
}

// ---------------------------------------------------------------------------
extern "C" void kernel_launch(void* const* d_in, const int* in_sizes, int n_in,
                              void* d_out, int out_size, void* d_ws, size_t ws_size,
                              hipStream_t stream) {
    const float* hidden = (const float*)d_in[0];
    const float* obs    = (const float*)d_in[1];
    const unsigned char* dones = (const unsigned char*)d_in[2];  // numpy bool (1B)
    const float* W_emb_t = (const float*)d_in[3];
    const float* b_emb_t = (const float*)d_in[4];
    const float* W_emb_a = (const float*)d_in[5];
    const float* b_emb_a = (const float*)d_in[6];
    const float* W_emb_c = (const float*)d_in[7];
    const float* b_emb_c = (const float*)d_in[8];
    const float* Wi_t = (const float*)d_in[9];
    const float* bi_t = (const float*)d_in[10];
    const float* Whrz_t = (const float*)d_in[11];
    const float* Whn_t = (const float*)d_in[12];
    const float* bhn_t = (const float*)d_in[13];
    const float* Wi_a = (const float*)d_in[14];
    const float* bi_a = (const float*)d_in[15];
    const float* Whrz_a = (const float*)d_in[16];
    const float* Whn_a = (const float*)d_in[17];
    const float* bhn_a = (const float*)d_in[18];
    const float* Wi_c = (const float*)d_in[19];
    const float* bi_c = (const float*)d_in[20];
    const float* Whrz_c = (const float*)d_in[21];
    const float* Whn_c = (const float*)d_in[22];
    const float* bhn_c = (const float*)d_in[23];
    const float* W_act = (const float*)d_in[24];
    const float* b_act = (const float*)d_in[25];
    const float* W_crit = (const float*)d_in[26];
    const float* b_crit = (const float*)d_in[27];
    const float* W_th = (const float*)d_in[28];
    const float* b_th = (const float*)d_in[29];

    char* ws = (char*)d_ws;
    unsigned short* wWc   = (unsigned short*)(ws + 0);
    unsigned short* wWhrz = (unsigned short*)(ws + 147456);
    unsigned short* wWhn  = (unsigned short*)(ws + 344064);
    unsigned short* wWth  = (unsigned short*)(ws + 442368);
    float*          wBc   = (float*)(ws + 475136);
    unsigned short* wObs  = (unsigned short*)(ws + 524288);
    unsigned short* wY    = (unsigned short*)(ws + 17825792);  // needs ~118.5MB total

    float* out = (float*)d_out;
    float* oHid  = out;             // [B,3H]      196608
    float* oAct  = out + 196608;    // [T,B,A]     786432
    float* oCrit = out + 983040;    // [T,B]       131072
    float* oTh   = out + 1114112;   // [T,B,H]     16777216

    k_wc<<<288, 256, 0, stream>>>(W_emb_t, W_emb_a, W_emb_c, Wi_t, Wi_a, Wi_c, wWc);
    k_bc<<<5, 256, 0, stream>>>(b_emb_t, b_emb_a, b_emb_c, Wi_t, Wi_a, Wi_c,
                                bi_t, bi_a, bi_c, wBc);
    k_swz<<<128, 256, 0, stream>>>(Whrz_t, wWhrz, 128, 256);
    k_swz<<<128, 256, 0, stream>>>(Whrz_a, wWhrz + 32768, 128, 256);
    k_swz<<<128, 256, 0, stream>>>(Whrz_c, wWhrz + 65536, 128, 256);
    k_swz<<<64, 256, 0, stream>>>(Whn_t, wWhn, 128, 128);
    k_swz<<<64, 256, 0, stream>>>(Whn_a, wWhn + 16384, 128, 128);
    k_swz<<<64, 256, 0, stream>>>(Whn_c, wWhn + 32768, 128, 128);
    k_swz<<<128, 256, 0, stream>>>(W_th, wWth, 128, 128);
    k_obs<<<2048, 256, 0, stream>>>(obs, wObs);

    k_scan<<<96, 128, 0, stream>>>(hidden, dones, bhn_t, bhn_a, bhn_c,
                                   wWc, wWhrz, wWhn, wBc, wObs, wY, oHid);

    k_th<<<256, 128, 0, stream>>>(wY /*head t*/, wWth, b_th, oTh);
    k_ac<<<512, 256, 0, stream>>>(wY + 16777216 /*head a*/,
                                  wY + 2 * 16777216 /*head c*/,
                                  W_act, b_act, W_crit, b_crit, oAct, oCrit);
}
